// MedicalTemporalLayer_79671643341454
// MI455X (gfx1250) — compile-verified, matrix operand order checked by reading, not run on hardware
//
#include <hip/hip_runtime.h>

typedef __attribute__((ext_vector_type(16))) _Float16 v16h;
typedef __attribute__((ext_vector_type(8)))  float    v8f;
typedef __attribute__((ext_vector_type(4)))  unsigned int v4u;
typedef __attribute__((ext_vector_type(8)))  int      v8i;
typedef __attribute__((ext_vector_type(4)))  int      v4i;

constexpr int kB = 16;
constexpr int kT = 2048;
constexpr int kF = 256;
constexpr int kD = 1024;
constexpr int kM = kB * kT;          // 32768 rows total
constexpr float kScale = 0.03125f;   // 1/sqrt(1024)
constexpr float kLogFreq = 0.00899447302f; // ln(10000)/1024

#if defined(__HIP_DEVICE_COMPILE__) && __has_builtin(__builtin_amdgcn_tensor_load_to_lds)
#define HAVE_TDM 1
#else
#define HAVE_TDM 0
#endif

// ---------------------------------------------------------------------------
// WMMA helpers (CDNA5 gfx1250, wave32): D = A(16x32 f16) * B(32x16 f16) + C(f32)
// ---------------------------------------------------------------------------
__device__ __forceinline__ v8f wmma_f16(v16h a, v16h b, v8f c) {
  // 8 args: (neg_a, A, neg_b, B, c_mod, C, reuse_a, reuse_b)
  return __builtin_amdgcn_wmma_f32_16x16x32_f16(false, a, false, b, (short)0, c,
                                                false, false);
}

__device__ __forceinline__ int frag_kbase(int v, int hi) {
  // 16-bit A/B operand K layout: VGPR v holds K pair starting at
  // (v>>2)*16 + (lane>=16 ? 8 : 0) + (v&3)*2
  return ((v >> 2) << 4) + (hi << 3) + ((v & 3) << 1);
}

// A fragment: 16x32 tile, rows row0..row0+15, K = k0..k0+31, from f32 row-major
__device__ __forceinline__ v16h load_a_f32(const float* __restrict__ src, int ld,
                                           int row0, int k0) {
  const int lane = threadIdx.x & 31;
  const int r = lane & 15, hi = lane >> 4;
  v16h a;
#pragma unroll
  for (int v = 0; v < 8; ++v) {
    const float* p = src + (size_t)(row0 + r) * ld + k0 + frag_kbase(v, hi);
    a[2 * v]     = (_Float16)p[0];
    a[2 * v + 1] = (_Float16)p[1];
  }
  return a;
}

// A fragment from f16 row-major
__device__ __forceinline__ v16h load_a_f16(const _Float16* __restrict__ src, int ld,
                                           int row0, int k0) {
  const int lane = threadIdx.x & 31;
  const int r = lane & 15, hi = lane >> 4;
  v16h a;
#pragma unroll
  for (int v = 0; v < 8; ++v) {
    const _Float16* p = src + (size_t)(row0 + r) * ld + k0 + frag_kbase(v, hi);
    a[2 * v]     = p[0];
    a[2 * v + 1] = p[1];
  }
  return a;
}

// B fragment: element(k,n) = src[(k0+k)*ld + n0+n], f16 row-major [K x N]
__device__ __forceinline__ v16h load_b_f16(const _Float16* __restrict__ src, int ld,
                                           int k0, int n0) {
  const int lane = threadIdx.x & 31;
  const int n = lane & 15, hi = lane >> 4;
  v16h b;
#pragma unroll
  for (int v = 0; v < 8; ++v) {
    const int kb = frag_kbase(v, hi);
    b[2 * v]     = src[(size_t)(k0 + kb) * ld + n0 + n];
    b[2 * v + 1] = src[(size_t)(k0 + kb + 1) * ld + n0 + n];
  }
  return b;
}

// B fragment of a TRANSPOSED source: element(k,n) = src[(n0+n)*ld + k0+k]
// (used for Q*K^T; works for global or LDS source — inlined so the address
//  space of __shared__ arrays is recovered and ds_load ops are emitted)
__device__ __forceinline__ v16h load_bT_f16(const _Float16* src, int ld,
                                            int n0, int k0) {
  const int lane = threadIdx.x & 31;
  const int n = lane & 15, hi = lane >> 4;
  v16h b;
#pragma unroll
  for (int v = 0; v < 8; ++v) {
    const _Float16* p = src + (size_t)(n0 + n) * ld + k0 + frag_kbase(v, hi);
    b[2 * v]     = p[0];
    b[2 * v + 1] = p[1];
  }
  return b;
}

// ---------------------------------------------------------------------------
// Tensor Data Mover: stage a [rows x kD] f16 tile (row-major, stride kD) from
// global memory into LDS. 2-D descriptor per CDNA5 ISA section 8.3/8.4:
//   group0: count=1 | lds_addr | global_addr(57b) | type=2
//   group1: data_size=1(2B) | tensor_dim0=kD | tensor_dim1=rows |
//           tile_dim0=kD | tile_dim1=rows | tile_dim2=0 | stride0=kD
// ---------------------------------------------------------------------------
#if HAVE_TDM
__device__ __forceinline__ void tdm_load_rows_f16(const _Float16* gptr,
                                                  unsigned lds_off, int rows) {
  const unsigned long long ga = (unsigned long long)(uintptr_t)gptr;
  v4u g0;
  g0.x = 1u;                                        // count=1 (valid user D#)
  g0.y = lds_off;                                   // LDS byte address
  g0.z = (unsigned)(ga & 0xFFFFFFFFu);              // global addr [31:0]
  g0.w = (unsigned)((ga >> 32) & 0x1FFFFFFu)        // global addr [56:32]
         | (2u << 30);                              // type=2 ("image")
  v8i g1;
  g1[0] = (int)(1u << 16);                          // wg_mask=0, data_size=1 (2B)
  g1[1] = (int)(((unsigned)kD & 0xFFFFu) << 16);    // tensor_dim0[15:0]
  g1[2] = (int)(((unsigned)kD >> 16) & 0xFFFFu)     // tensor_dim0[31:16]
          | (int)(((unsigned)rows & 0xFFFFu) << 16);// tensor_dim1[15:0]
  g1[3] = (int)(((unsigned)rows >> 16) & 0xFFFFu)   // tensor_dim1[31:16]
          | (int)(((unsigned)kD & 0xFFFFu) << 16);  // tile_dim0
  g1[4] = (int)((unsigned)rows & 0xFFFFu);          // tile_dim1 (tile_dim2=0)
  g1[5] = kD;                                       // tensor_dim0_stride[31:0]
  g1[6] = 0;                                        // stride0 hi / stride1 lo
  g1[7] = 0;                                        // stride1 hi
  v4i z4 = {0, 0, 0, 0};
#if defined(__clang_major__) && __clang_major__ >= 23
  v8i z8 = {0, 0, 0, 0, 0, 0, 0, 0};
  __builtin_amdgcn_tensor_load_to_lds(g0, g1, z4, z4, z8, 0);
#else
  __builtin_amdgcn_tensor_load_to_lds(g0, g1, z4, z4, 0);
#endif
}
#endif

// ---------------------------------------------------------------------------
// Kernel 0: f32 -> f16 weight conversion
// ---------------------------------------------------------------------------
__global__ void k_cvt_f16(const float* __restrict__ src, _Float16* __restrict__ dst,
                          int n) {
  int i = blockIdx.x * blockDim.x + threadIdx.x;
  if (i < n) dst[i] = (_Float16)src[i];
}

// ---------------------------------------------------------------------------
// Kernel 1: positional time encoding of inter-event deltas, f16, padded to T
// rows per batch (row T-1 is zero; masked to -inf in attention logits).
// ---------------------------------------------------------------------------
__global__ void k_time_enc(const float* __restrict__ times,
                           _Float16* __restrict__ te) {
  size_t idx = (size_t)blockIdx.x * blockDim.x + threadIdx.x;
  int j = (int)(idx % kD);
  size_t row = idx / kD;            // b*kT + s
  int s = (int)(row % kT);
  int b = (int)(row / kT);
  float val = 0.0f;
  if (s < kT - 1) {
    float dt = times[b * kT + s + 1] - times[b * kT + s];
    int jj = (j < kD / 2) ? j : j - kD / 2;
    float freq = __expf(-(float)(2 * jj) * kLogFreq);
    float ang = dt * freq;
    val = (j < kD / 2) ? __sinf(ang) : __cosf(ang);
  }
  te[idx] = (_Float16)val;
}

// ---------------------------------------------------------------------------
// Kernel 2: value_enc = values(f32) @ value_kernel(f16)  ->  f32 [kM, kD]
// One wave per 16x16 C tile, K = 256.
// ---------------------------------------------------------------------------
__global__ void k_gemm_valenc(const float* __restrict__ values,
                              const _Float16* __restrict__ vk,
                              float* __restrict__ out) {
  const int wid = blockIdx.x * (blockDim.x >> 5) + (threadIdx.x >> 5);
  const int tn = wid & (kD / 16 - 1);
  const int tm = wid / (kD / 16);
  const int row0 = tm * 16, n0 = tn * 16;
  v8f acc = {};
  for (int k0 = 0; k0 < kF; k0 += 32) {
    __builtin_prefetch(values + (size_t)row0 * kF + k0 + 32, 0, 1);
    v16h a = load_a_f32(values, kF, row0, k0);
    v16h b = load_b_f16(vk, kD, k0, n0);
    acc = wmma_f16(a, b, acc);
  }
  const int lane = threadIdx.x & 31;
  const int col = n0 + (lane & 15);
  const int rbase = row0 + 8 * (lane >> 4);
#pragma unroll
  for (int v = 0; v < 8; ++v) out[(size_t)(rbase + v) * kD + col] = acc[v];
}

// ---------------------------------------------------------------------------
// Kernel 3: tw = time_enc(f16) @ attention_w(f16)  ->  f16 [kM, kD]
// Padded time_enc rows are zero, so padded tw rows come out zero.
// ---------------------------------------------------------------------------
__global__ void k_gemm_tw(const _Float16* __restrict__ te,
                          const _Float16* __restrict__ aw,
                          _Float16* __restrict__ tw) {
  const int wid = blockIdx.x * (blockDim.x >> 5) + (threadIdx.x >> 5);
  const int tn = wid & (kD / 16 - 1);
  const int tm = wid / (kD / 16);
  const int row0 = tm * 16, n0 = tn * 16;
  v8f acc = {};
  for (int k0 = 0; k0 < kD; k0 += 32) {
    v16h a = load_a_f16(te, kD, row0, k0);
    v16h b = load_b_f16(aw, kD, k0, n0);
    acc = wmma_f16(a, b, acc);
  }
  const int lane = threadIdx.x & 31;
  const int col = n0 + (lane & 15);
  const int rbase = row0 + 8 * (lane >> 4);
#pragma unroll
  for (int v = 0; v < 8; ++v)
    tw[(size_t)(rbase + v) * kD + col] = (_Float16)acc[v];
}

// ---------------------------------------------------------------------------
// Kernel 4: flash attention.
//   One block (8 waves) per 16-row Q tile; each wave owns a 128-wide D slice.
//   Per 32-step of s: wave0 TDM-stages the 32x1024 f16 K-tile (64KB) into LDS
//   (tensor_load_to_lds + s_wait_tensorcnt), all waves build Q*K^T B-operands
//   from LDS, run online softmax, reshape P (C-layout -> A-layout) through an
//   8KB staging region overlaid on the head of the K-tile, then P*V WMMA.
// ---------------------------------------------------------------------------
__global__ void k_flash(const float* __restrict__ value_enc,
                        const _Float16* __restrict__ time_enc,
                        const _Float16* __restrict__ twv,
                        float* __restrict__ head) {
  __shared__ _Float16 smem[32 * 1024];       // 64KB K-tile stage; pbuf overlays
  const int warp = threadIdx.x >> 5;
  const int lane = threadIdx.x & 31;
  const int r = lane & 15, hi = lane >> 4;
  const int row0 = blockIdx.x * 16;          // global Q row (batch-aligned)
  const int b = row0 / kT;
  const int ds = warp * 128;                 // this wave's D slice
  const _Float16* te = time_enc + (size_t)b * kT * kD;
  const _Float16* tw = twv + (size_t)b * kT * kD;
  _Float16* pb = smem + warp * (16 * 32);    // per-wave 1KB P staging

  float mrow[8], lrow[8];
  v8f o[8];
  v8f zero = {};
#pragma unroll
  for (int v = 0; v < 8; ++v) { mrow[v] = -1e30f; lrow[v] = 0.0f; }
#pragma unroll
  for (int t = 0; t < 8; ++t) o[t] = zero;

  for (int s0 = 0; s0 < kT; s0 += 32) {
    // ---- stage K-tile rows [s0, s0+32) x kD into LDS ----
#if HAVE_TDM
    if (threadIdx.x < 32) {
      tdm_load_rows_f16(te + (size_t)s0 * kD,
                        (unsigned)(uintptr_t)(void*)smem, 32);
      __builtin_amdgcn_s_wait_tensorcnt(0);
    }
#else
    {
      const uint4* src = (const uint4*)(te + (size_t)s0 * kD);
      uint4* dst = (uint4*)smem;
      for (int i = threadIdx.x; i < (32 * kD) / 8; i += 256) dst[i] = src[i];
    }
#endif
    __syncthreads();

    // ---- S = Q * K^T over full D (K operand from LDS) ----
    v8f slo = zero, shi = zero;
    for (int k0 = 0; k0 < kD; k0 += 32) {
      v16h q  = load_a_f32(value_enc, kD, row0, k0);
      v16h k1 = load_bT_f16(smem, kD, 0, k0);
      v16h k2 = load_bT_f16(smem, kD, 16, k0);
      slo = wmma_f16(q, k1, slo);
      shi = wmma_f16(q, k2, shi);
    }
    __syncthreads();   // all waves done with K-tile before pbuf overlay write

    // scale + mask padded column (s == kT-1 does not exist in reference)
    const int clo = s0 + r, chi = clo + 16;
#pragma unroll
    for (int v = 0; v < 8; ++v) {
      slo[v] = (clo < kT - 1) ? slo[v] * kScale : -1e30f;
      shi[v] = (chi < kT - 1) ? shi[v] * kScale : -1e30f;
    }
    // ---- online softmax: row max over 32 cols (16-lane half reduce) ----
    float bm[8];
#pragma unroll
    for (int v = 0; v < 8; ++v) bm[v] = fmaxf(slo[v], shi[v]);
#pragma unroll
    for (int off = 1; off <= 8; off <<= 1)
#pragma unroll
      for (int v = 0; v < 8; ++v)
        bm[v] = fmaxf(bm[v], __shfl_xor(bm[v], off, 32));
    float alpha[8], ps[8];
#pragma unroll
    for (int v = 0; v < 8; ++v) {
      float mn = fmaxf(mrow[v], bm[v]);
      alpha[v] = __expf(mrow[v] - mn);
      mrow[v] = mn;
      slo[v] = __expf(slo[v] - mn);
      shi[v] = __expf(shi[v] - mn);
      ps[v] = slo[v] + shi[v];
    }
#pragma unroll
    for (int off = 1; off <= 8; off <<= 1)
#pragma unroll
      for (int v = 0; v < 8; ++v) ps[v] += __shfl_xor(ps[v], off, 32);
#pragma unroll
    for (int v = 0; v < 8; ++v) lrow[v] = lrow[v] * alpha[v] + ps[v];
#pragma unroll
    for (int t = 0; t < 8; ++t)
#pragma unroll
      for (int v = 0; v < 8; ++v) o[t][v] *= alpha[v];

    // ---- stash P (C layout) into LDS as row-major 16x32 f16 ----
#pragma unroll
    for (int v = 0; v < 8; ++v) {
      pb[(v + 8 * hi) * 32 + r]      = (_Float16)slo[v];
      pb[(v + 8 * hi) * 32 + 16 + r] = (_Float16)shi[v];
    }
    __syncthreads();
    // reload as WMMA A operand (16x32, K=32)
    v16h pf;
#pragma unroll
    for (int v = 0; v < 8; ++v) {
      const int kb = frag_kbase(v, hi);
      pf[2 * v]     = pb[r * 32 + kb];
      pf[2 * v + 1] = pb[r * 32 + kb + 1];
    }
    __syncthreads();   // P consumed before next iteration's TDM overwrite

    // ---- O += P * V for this wave's 128-wide D slice ----
#pragma unroll
    for (int t = 0; t < 8; ++t) {
      v16h vf = load_b_f16(tw + (size_t)s0 * kD, kD, 0, ds + t * 16);
      o[t] = wmma_f16(pf, vf, o[t]);
    }
  }

#pragma unroll
  for (int v = 0; v < 8; ++v) lrow[v] = 1.0f / lrow[v];
  const int rbase = row0 + 8 * hi;
#pragma unroll
  for (int t = 0; t < 8; ++t) {
    const int col = ds + t * 16 + r;
#pragma unroll
    for (int v = 0; v < 8; ++v)
      head[(size_t)(rbase + v) * kD + col] = o[t][v] * lrow[v];
  }
}

// ---------------------------------------------------------------------------
// Kernel 5: gate = sigmoid([ve, head] @ gate_w); out = g*ve + (1-g)*head
// K = 2048 split into the ve half and head half of gate_w.
// ---------------------------------------------------------------------------
__global__ void k_gate(const float* __restrict__ ve,
                       const float* __restrict__ head,
                       const _Float16* __restrict__ gw,
                       float* __restrict__ out) {
  const int wid = blockIdx.x * (blockDim.x >> 5) + (threadIdx.x >> 5);
  const int tn = wid & (kD / 16 - 1);
  const int tm = wid / (kD / 16);
  const int row0 = tm * 16, n0 = tn * 16;
  v8f acc = {};
  for (int k0 = 0; k0 < kD; k0 += 32) {
    v16h a = load_a_f32(ve, kD, row0, k0);
    v16h b = load_b_f16(gw, kD, k0, n0);
    acc = wmma_f16(a, b, acc);
  }
  const _Float16* gw_hi = gw + (size_t)kD * kD;
  for (int k0 = 0; k0 < kD; k0 += 32) {
    v16h a = load_a_f32(head, kD, row0, k0);
    v16h b = load_b_f16(gw_hi, kD, k0, n0);
    acc = wmma_f16(a, b, acc);
  }
  const int lane = threadIdx.x & 31;
  const int col = n0 + (lane & 15);
  const int rbase = row0 + 8 * (lane >> 4);
#pragma unroll
  for (int v = 0; v < 8; ++v) {
    const size_t idx = (size_t)(rbase + v) * kD + col;
    const float g = 1.0f / (1.0f + __expf(-acc[v]));
    out[idx] = g * ve[idx] + (1.0f - g) * head[idx];
  }
}

// ---------------------------------------------------------------------------
extern "C" void kernel_launch(void* const* d_in, const int* in_sizes, int n_in,
                              void* d_out, int out_size, void* d_ws, size_t ws_size,
                              hipStream_t stream) {
  (void)in_sizes; (void)n_in; (void)out_size; (void)ws_size;
  const float* values = (const float*)d_in[0];   // [B,T,F]
  const float* times  = (const float*)d_in[1];   // [B,T]
  const float* vkf    = (const float*)d_in[2];   // [F,D]
  const float* awf    = (const float*)d_in[3];   // [D,D]
  const float* gwf    = (const float*)d_in[4];   // [2D,D]
  float* out = (float*)d_out;                    // [B,T,D]

  // workspace layout
  char* w = (char*)d_ws;
  float*    value_enc = (float*)w;     w += (size_t)kM * kD * sizeof(float);
  _Float16* te16      = (_Float16*)w;  w += (size_t)kM * kD * sizeof(_Float16);
  _Float16* tw16      = (_Float16*)w;  w += (size_t)kM * kD * sizeof(_Float16);
  float*    head      = (float*)w;     w += (size_t)kM * kD * sizeof(float);
  _Float16* vk16      = (_Float16*)w;  w += (size_t)kF * kD * sizeof(_Float16);
  _Float16* aw16      = (_Float16*)w;  w += (size_t)kD * kD * sizeof(_Float16);
  _Float16* gw16      = (_Float16*)w;  w += (size_t)2 * kD * kD * sizeof(_Float16);

  k_cvt_f16<<<(kF * kD + 255) / 256, 256, 0, stream>>>(vkf, vk16, kF * kD);
  k_cvt_f16<<<(kD * kD + 255) / 256, 256, 0, stream>>>(awf, aw16, kD * kD);
  k_cvt_f16<<<(2 * kD * kD + 255) / 256, 256, 0, stream>>>(gwf, gw16, 2 * kD * kD);

  k_time_enc<<<(int)(((size_t)kM * kD) / 256), 256, 0, stream>>>(times, te16);

  const int tiles = (kM / 16) * (kD / 16);   // 131072 16x16 tiles
  k_gemm_valenc<<<tiles / 8, 256, 0, stream>>>(values, vk16, value_enc);
  k_gemm_tw<<<tiles / 8, 256, 0, stream>>>(te16, aw16, tw16);

  k_flash<<<kM / 16, 256, 0, stream>>>(value_enc, te16, tw16, head);

  k_gate<<<tiles / 8, 256, 0, stream>>>(value_enc, head, gw16, out);
}
